// BrainGCN_35321811042312
// MI455X (gfx1250) — matrix operand order repaired and checked by reading, not compiled, not run on hardware
//
#include <hip/hip_runtime.h>
#include <hip/hip_bf16.h>

typedef __attribute__((ext_vector_type(16))) _Float16 v16h;
typedef __attribute__((ext_vector_type(8)))  float    v8f;

#define NN   50000
#define EE   800000
#define GG   128
#define HID  64
#define EPSV 1e-5f

// ---------------------------------------------------------------- utilities
__global__ void k_fill(float* __restrict__ p, float v, int n) {
    int i = blockIdx.x * blockDim.x + threadIdx.x;
    if (i < n) p[i] = v;
}

// deg[dst] += w  (deg pre-filled with 1.0 for the self loop)
__global__ void k_deg(const int* __restrict__ ei, const float* __restrict__ w,
                      float* __restrict__ deg) {
    int e = blockIdx.x * blockDim.x + threadIdx.x;
    if (e < EE) atomicAdd(&deg[ei[EE + e]], w[e]);
}

__global__ void k_dinv(const float* __restrict__ deg, float* __restrict__ dinv) {
    int i = blockIdx.x * blockDim.x + threadIdx.x;
    if (i < NN) {
        float d = deg[i];
        dinv[i] = (d > 0.f) ? rsqrtf(fmaxf(d, 1e-12f)) : 0.f;
    }
}

// ------------------------------------------------- layer-1 projection (K=5)
__global__ void k_proj1(const float* __restrict__ x, const float* __restrict__ W1,
                        float* __restrict__ P) {
    int t = blockIdx.x * blockDim.x + threadIdx.x;   // (node, feat)
    if (t >= NN * HID) return;
    int i = t >> 6, f = t & 63;
    const float* xr = x + i * 5;
    float acc = 0.f;
#pragma unroll
    for (int k = 0; k < 5; ++k) acc += xr[k] * W1[k * HID + f];
    P[t] = acc;
}

// ----------------------------------------------- edge scatter (agg pre-zeroed)
// 64 threads per edge -> coalesced 64-float message rows, f32 global atomics.
__global__ void k_scatter(const int* __restrict__ ei, const float* __restrict__ w,
                          const float* __restrict__ dinv, const float* __restrict__ P,
                          float* __restrict__ agg) {
    long long t = (long long)blockIdx.x * blockDim.x + threadIdx.x;
    long long e = t >> 6;
    if (e >= EE) return;
    int f = (int)(t & 63);
    int s = ei[e], d = ei[EE + e];
    float norm = dinv[s] * w[e] * dinv[d];
    atomicAdd(&agg[(long long)d * HID + f], norm * P[(long long)s * HID + f]);
}

// --------------------------- fused self-loop + bias + batchnorm(eval) + ReLU
__global__ void k_bnrelu(const float* __restrict__ agg, const float* __restrict__ P,
                         const float* __restrict__ dinv, const float* __restrict__ b,
                         const float* __restrict__ gamma, const float* __restrict__ beta,
                         const float* __restrict__ mean, const float* __restrict__ var,
                         float* __restrict__ out) {
    long long t = (long long)blockIdx.x * blockDim.x + threadIdx.x;
    if (t >= (long long)NN * HID) return;
    int i = (int)(t >> 6), f = (int)(t & 63);
    float di = dinv[i];
    float v = agg[t] + di * di * P[t] + b[f];
    float h = (v - mean[f]) * rsqrtf(var[f] + EPSV) * gamma[f] + beta[f];
    out[t] = fmaxf(h, 0.f);
}

// ---------------------------------------------------------- W2 -> f16 (once)
__global__ void k_w2half(const float* __restrict__ W, _Float16* __restrict__ Wh) {
    int i = blockIdx.x * blockDim.x + threadIdx.x;
    if (i < HID * HID) Wh[i] = (_Float16)W[i];
}

// -------------------------------------- P = H[N,64] @ W2[64,64] via WMMA f16
// One wave per 16x16 output tile, K=64 -> two v_wmma_f32_16x16x32_f16.
// N = 50000 = 3125 * 16 exactly, so every wave is full (EXEC all-1s).
__global__ void __launch_bounds__(256)
k_gemm_wmma(const float* __restrict__ H, const _Float16* __restrict__ Wh,
            float* __restrict__ P) {
    int wave = (int)((blockIdx.x * blockDim.x + threadIdx.x) >> 5);
    int lane = threadIdx.x & 31;
    const int numTiles = (NN / 16) * (HID / 16);     // 12500
    if (wave >= numTiles) return;                    // wave-uniform exit
    int rb = wave >> 2, cb = wave & 3;
    int m0 = rb * 16, n0 = cb * 16;
    int hi   = lane >> 4;                            // 0 | 1
    int row  = m0 + (lane & 15);                     // A row for this lane
    int col  = n0 + (lane & 15);                     // B/C/D column for this lane
    v8f c = {};
#pragma unroll
    for (int k0 = 0; k0 < HID; k0 += 32) {
        int ka = k0 + (hi << 4);                     // A: K=0..15 / 16..31
        int kb = k0 + (hi << 4);                     // B: K=0..15 / 16..31
        v16h a, b;
        const float* hr = H + (long long)row * HID;
#pragma unroll
        for (int i = 0; i < 8; ++i) {
            a[i]     = (_Float16)hr[ka + i];
            a[8 + i] = (_Float16)hr[ka + 8 + i];
        }
#pragma unroll
        for (int i = 0; i < 16; ++i)
            b[i] = Wh[(kb + i) * HID + col];
        c = __builtin_amdgcn_wmma_f32_16x16x32_f16(
                /*neg_a=*/false, a, /*neg_b=*/false, b,
                /*c_mod=*/(short)0, c, /*reuse_a=*/false, /*reuse_b=*/false);
    }
    int rbase = m0 + (hi << 3);
#pragma unroll
    for (int r = 0; r < 8; ++r)
        P[(long long)(rbase + r) * HID + col] = c[r];
}

// ------------------------------------------------------- global mean pooling
__global__ void k_pool(const float* __restrict__ h, const int* __restrict__ batch,
                       float* __restrict__ sums, float* __restrict__ cnt) {
    long long t = (long long)blockIdx.x * blockDim.x + threadIdx.x;
    if (t >= (long long)NN * HID) return;
    int i = (int)(t >> 6), f = (int)(t & 63);
    int g = batch[i];
    atomicAdd(&sums[g * HID + f], h[t]);
    if (f == 0) atomicAdd(&cnt[g], 1.0f);
}

// ---------------------------------------------------- MLP head (G=128 tiny)
__global__ void k_head(const float* __restrict__ sums, const float* __restrict__ cnt,
                       const float* __restrict__ l1W, const float* __restrict__ l1b,
                       const float* __restrict__ l2W, const float* __restrict__ l2b,
                       float* __restrict__ out) {
    int g = threadIdx.x;
    if (g >= GG) return;
    float inv = 1.0f / fmaxf(cnt[g], 1.0f);
    const float* sg = sums + g * HID;
    float z[32];
#pragma unroll 4
    for (int j = 0; j < 32; ++j) {
        float acc = l1b[j];
        for (int k = 0; k < HID; ++k) acc += sg[k] * inv * l1W[k * 32 + j];
        z[j] = fmaxf(acc, 0.f);
    }
    float o = l2b[0];
#pragma unroll
    for (int j = 0; j < 32; ++j) o += z[j] * l2W[j];
    out[g] = o;
}

extern "C" void kernel_launch(void* const* d_in, const int* in_sizes, int n_in,
                              void* d_out, int out_size, void* d_ws, size_t ws_size,
                              hipStream_t stream) {
    const float* x    = (const float*)d_in[0];
    const int*   ei   = (const int*)  d_in[1];   // [2,E] flat: src=[0..E), dst=[E..2E)
    const float* ew   = (const float*)d_in[2];
    const int*   batc = (const int*)  d_in[3];
    const float* W1   = (const float*)d_in[4];
    const float* b1   = (const float*)d_in[5];
    const float* W2   = (const float*)d_in[6];
    const float* b2   = (const float*)d_in[7];
    const float* g1   = (const float*)d_in[8];
    const float* be1  = (const float*)d_in[9];
    const float* m1   = (const float*)d_in[10];
    const float* v1   = (const float*)d_in[11];
    const float* g2   = (const float*)d_in[12];
    const float* be2  = (const float*)d_in[13];
    const float* m2   = (const float*)d_in[14];
    const float* v2   = (const float*)d_in[15];
    const float* l1W  = (const float*)d_in[16];
    const float* l1b  = (const float*)d_in[17];
    const float* l2W  = (const float*)d_in[18];
    const float* l2b  = (const float*)d_in[19];
    float* out = (float*)d_out;

    // ---- workspace carve-up (floats)
    float* ws   = (float*)d_ws;
    float* proj = ws;                       // N*64
    float* agg  = proj + (size_t)NN * HID;  // N*64
    float* hbuf = agg  + (size_t)NN * HID;  // N*64
    float* deg  = hbuf + (size_t)NN * HID;  // N
    float* dinv = deg  + NN;                // N
    float* sums = dinv + NN;                // G*64
    float* cnt  = sums + (size_t)GG * HID;  // G
    _Float16* W2h = (_Float16*)(cnt + GG);  // 64*64 halfs

    const int T = 256;
    const long long NF = (long long)NN * HID;
    dim3 gNF((unsigned)((NF + T - 1) / T));
    dim3 gEF((unsigned)(((long long)EE * HID + T - 1) / T));
    dim3 gN((NN + T - 1) / T);
    dim3 gE((EE + T - 1) / T);

    // degrees + normalization
    k_fill<<<gN, T, 0, stream>>>(deg, 1.0f, NN);                       // self-loop wt
    k_fill<<<gNF, T, 0, stream>>>(agg, 0.0f, (int)NF);
    k_fill<<<(GG * (HID + 1) + T - 1) / T, T, 0, stream>>>(sums, 0.0f, GG * (HID + 1));
    k_deg<<<gE, T, 0, stream>>>(ei, ew, deg);
    k_dinv<<<gN, T, 0, stream>>>(deg, dinv);

    // layer 1: project (K=5), scatter, bn+relu
    k_proj1<<<gNF, T, 0, stream>>>(x, W1, proj);
    k_scatter<<<gEF, T, 0, stream>>>(ei, ew, dinv, proj, agg);
    k_bnrelu<<<gNF, T, 0, stream>>>(agg, proj, dinv, b1, g1, be1, m1, v1, hbuf);

    // layer 2: WMMA GEMM (h1 @ W2), scatter, bn+relu
    k_fill<<<gNF, T, 0, stream>>>(agg, 0.0f, (int)NF);
    k_w2half<<<(HID * HID + T - 1) / T, T, 0, stream>>>(W2, W2h);
    {
        const int waves = (NN / 16) * (HID / 16);          // 12500
        dim3 gW((waves * 32 + T - 1) / T);                 // 8 waves / block
        k_gemm_wmma<<<gW, T, 0, stream>>>(hbuf, W2h, proj);
    }
    k_scatter<<<gEF, T, 0, stream>>>(ei, ew, dinv, proj, agg);
    k_bnrelu<<<gNF, T, 0, stream>>>(agg, proj, dinv, b2, g2, be2, m2, v2, hbuf);

    // pooling + head
    k_pool<<<gNF, T, 0, stream>>>(hbuf, batc, sums, cnt);
    k_head<<<1, 128, 0, stream>>>(sums, cnt, l1W, l1b, l2W, l2b, out);
}
